// RandomizedMessagePassingLayer_43181601194858
// MI455X (gfx1250) — compile-verified, matrix operand order checked by reading, not run on hardware
//
#include <hip/hip_runtime.h>
#include <hip/hip_bf16.h>
#include <math.h>

// ---------------------------------------------------------------------------
// RandomizedMessagePassingLayer on MI455X (gfx1250, wave32)
//
// x2 = swish( (adj @ swish((adj @ x) @ W + b)) @ W + b )
//
// adj is 0/1 with avg degree 32 (0.2% dense). Build CSR in one 1 GiB
// streaming pass, do both hop aggregations as L2-resident exact-f32
// gather-sums, and run the two [M,128]x[128,128] feature GEMMs on the WMMA
// pipe (v_wmma_f32_16x16x32_f16, f32 accumulate) with fused bias+swish.
// W is pre-packed once into an f16 fragment-ready layout so each lane's
// B fragment is two contiguous global_load_b128.
// ---------------------------------------------------------------------------

typedef __attribute__((ext_vector_type(16))) _Float16 v16h;
typedef __attribute__((ext_vector_type(8)))  float    v8f;

#define NNODES 16384
#define UNITS  128
#define CAP    128   // max neighbors kept per row (mean 32, 128 is ~17 sigma)

// ---------------------------------------------------------------------------
// Kernel 1: scan one adjacency row per block, compact nonzero column indices.
// Streams 1 GiB of adj exactly once (float4 loads, fully coalesced).
// ---------------------------------------------------------------------------
__global__ void rmp_build_csr(const float* __restrict__ adj,
                              int* __restrict__ deg,
                              int* __restrict__ idx) {
    __shared__ int cnt;
    const int row = blockIdx.x;
    if (threadIdx.x == 0) cnt = 0;
    __syncthreads();

    const float4* arow = (const float4*)(adj + (size_t)row * NNODES);
    int* ridx = idx + (size_t)row * CAP;

    for (int c4 = threadIdx.x; c4 < NNODES / 4; c4 += blockDim.x) {
        float4 v = arow[c4];
        int base = c4 * 4;
        if (v.x != 0.0f) { int p = atomicAdd(&cnt, 1); if (p < CAP) ridx[p] = base + 0; }
        if (v.y != 0.0f) { int p = atomicAdd(&cnt, 1); if (p < CAP) ridx[p] = base + 1; }
        if (v.z != 0.0f) { int p = atomicAdd(&cnt, 1); if (p < CAP) ridx[p] = base + 2; }
        if (v.w != 0.0f) { int p = atomicAdd(&cnt, 1); if (p < CAP) ridx[p] = base + 3; }
    }
    __syncthreads();
    if (threadIdx.x == 0) deg[row] = (cnt < CAP) ? cnt : CAP;
}

// ---------------------------------------------------------------------------
// Kernel 2: pre-pack W (f32 row-major [128,128]) into the exact per-lane
// v16h B-fragment layout of v_wmma_f32_16x16x32_f16 (ISA 7.12.2):
//   Wp[((t*4 + kc)*32 + lane) * 16 + j] = W[k][16*t + (lane&15)]
//   with k = 32*kc + (j<8 ? j : j+8) + 8*(lane>>4)
// 8 column-tiles x 4 K-chunks x 32 lanes x 16 halves = 32 KB, L0/L2-resident.
// ---------------------------------------------------------------------------
__global__ void rmp_pack_w(const float* __restrict__ W,
                           _Float16* __restrict__ Wp) {
    const int f = blockIdx.x * blockDim.x + threadIdx.x;   // 0..16383
    const int j    = f & 15;
    const int lane = (f >> 4) & 31;
    const int kc   = (f >> 9) & 3;
    const int t    = f >> 11;
    const int lo = lane & 15, hi = lane >> 4;
    const int n = t * 16 + lo;
    const int k = kc * 32 + ((j < 8) ? j : (j + 8)) + 8 * hi;
    Wp[f] = (_Float16)W[k * UNITS + n];
}

// ---------------------------------------------------------------------------
// Kernel 3: neigh[row,:] = sum_{j in nbrs(row)} src[j,:]
// One 128-thread block (4 waves) per row; thread u owns feature u.
// src (8 MB) is fully L2-resident; gathers of 512 B rows are coalesced.
// ---------------------------------------------------------------------------
__global__ void rmp_aggregate(const float* __restrict__ src,
                              const int* __restrict__ deg,
                              const int* __restrict__ idx,
                              float* __restrict__ dst) {
    const int row = blockIdx.x;
    const int u   = threadIdx.x;           // 0..127
    const int* ridx = idx + (size_t)row * CAP;
    const int d = deg[row];

    float acc = 0.0f;
    for (int i = 0; i < d; ++i) {
        int j = ridx[i];
        if (i + 1 < d) {
            // gfx1250 global_prefetch_b8 of the next gathered row
            __builtin_prefetch(src + (size_t)ridx[i + 1] * UNITS + u, 0, 0);
        }
        acc += src[(size_t)j * UNITS + u];
    }
    dst[(size_t)row * UNITS + u] = acc;
}

// ---------------------------------------------------------------------------
// Kernel 4: out = swish(A @ W + b) via v_wmma_f32_16x16x32_f16.
// A: [M,128] f32 row-major; Wp: pre-packed f16 B fragments; b: [128].
// Block = 256 threads = 8 waves; wave w computes the 16x16 tile at
// rows [16*blockIdx.x, +16), cols [16*w, +16). K = 128 in 4 chunks of 32.
//
// ISA 7.12.2 16-bit fragment layouts (wave32), lo = lane&15, hi = lane>>4:
//   A (16x32, M x K): element j of v16h -> A[lo][ k0 + (j<8 ? j : j+8) + 8*hi ]
//   C/D (16x16 f32):  vgpr r, lane      -> D[ r + 8*hi ][ lo ]
// ---------------------------------------------------------------------------
__global__ void rmp_gemm_swish_wmma(const float* __restrict__ A,
                                    const v16h* __restrict__ Wp,
                                    const float* __restrict__ bias,
                                    float* __restrict__ out) {
    const int lane = threadIdx.x & 31;
    const int wave = threadIdx.x >> 5;     // 0..7 -> output column tile
    const int lo   = lane & 15;
    const int hi   = lane >> 4;

    const int m = blockIdx.x * 16 + lo;    // A row this lane supplies
    const int n = wave * 16 + lo;          // output column this lane owns

    const float* arow  = A + (size_t)m * UNITS;
    const v16h*  wtile = Wp + (size_t)wave * 128 + lane;  // + kc*32 per chunk

    v8f acc = {};

    #pragma unroll
    for (int kc = 0; kc < 4; ++kc) {
        const int k0 = kc * 32;

        // ---- A fragment: two contiguous runs of 8 floats -> f16 ----
        const int ab = k0 + 8 * hi;
        float4 p0 = *(const float4*)(arow + ab);
        float4 p1 = *(const float4*)(arow + ab + 4);
        float4 q0 = *(const float4*)(arow + ab + 16);
        float4 q1 = *(const float4*)(arow + ab + 20);

        v16h afrag;
        afrag[0]  = (_Float16)p0.x; afrag[1]  = (_Float16)p0.y;
        afrag[2]  = (_Float16)p0.z; afrag[3]  = (_Float16)p0.w;
        afrag[4]  = (_Float16)p1.x; afrag[5]  = (_Float16)p1.y;
        afrag[6]  = (_Float16)p1.z; afrag[7]  = (_Float16)p1.w;
        afrag[8]  = (_Float16)q0.x; afrag[9]  = (_Float16)q0.y;
        afrag[10] = (_Float16)q0.z; afrag[11] = (_Float16)q0.w;
        afrag[12] = (_Float16)q1.x; afrag[13] = (_Float16)q1.y;
        afrag[14] = (_Float16)q1.z; afrag[15] = (_Float16)q1.w;

        // ---- B fragment: one 32-byte contiguous load (2x b128), no cvt ----
        v16h bfrag = wtile[k0];

        // D = A x B + C  (f32 accumulate on the matrix pipe)
        acc = __builtin_amdgcn_wmma_f32_16x16x32_f16(
                  /*neg_a=*/false, afrag, /*neg_b=*/false, bfrag,
                  /*c_mod=*/(short)0, acc, /*reuse_a=*/false, /*reuse_b=*/false);
    }

    // ---- epilogue: bias + swish, scatter per the C/D layout ----
    const float bn = bias[n];
    #pragma unroll
    for (int r = 0; r < 8; ++r) {
        const int mrow = blockIdx.x * 16 + r + 8 * hi;
        float v = acc[r] + bn;
        float s = v * (1.0f / (1.0f + __expf(-v)));
        out[(size_t)mrow * UNITS + n] = s;
    }
}

// ---------------------------------------------------------------------------
// Host-side orchestration (graph-capture safe: launches only).
// ---------------------------------------------------------------------------
extern "C" void kernel_launch(void* const* d_in, const int* in_sizes, int n_in,
                              void* d_out, int out_size, void* d_ws, size_t ws_size,
                              hipStream_t stream) {
    const float* x   = (const float*)d_in[0];   // [N, U]
    const float* adj = (const float*)d_in[1];   // [N, N]
    const float* W   = (const float*)d_in[2];   // [U, U]
    const float* b   = (const float*)d_in[3];   // [U]
    float* x2        = (float*)d_out;           // [N, U]

    // Workspace layout
    char* ws = (char*)d_ws;
    const size_t degBytes  = (size_t)NNODES * sizeof(int);               // 64 KB
    const size_t idxBytes  = (size_t)NNODES * CAP * sizeof(int);         // 8 MB
    const size_t featBytes = (size_t)NNODES * UNITS * sizeof(float);     // 8 MB
    int*      deg   = (int*)ws;
    int*      idx   = (int*)(ws + degBytes);
    float*    neigh = (float*)(ws + degBytes + idxBytes);
    float*    h1    = (float*)(ws + degBytes + idxBytes + featBytes);
    _Float16* Wp    = (_Float16*)(ws + degBytes + idxBytes + 2 * featBytes); // 32 KB

    // 0. pre-pack W into WMMA B-fragment layout (32 KB, L0/L2-resident)
    rmp_pack_w<<<64, 256, 0, stream>>>(W, Wp);

    // 1. One streaming pass over adj -> CSR (the only HBM-bound step)
    rmp_build_csr<<<NNODES, 256, 0, stream>>>(adj, deg, idx);

    // 2. hop 1 aggregate (L2-resident gather-sum, exact f32)
    rmp_aggregate<<<NNODES, UNITS, 0, stream>>>(x, deg, idx, neigh);

    // 3. x1 = swish(neigh @ W + b) on the WMMA pipe
    rmp_gemm_swish_wmma<<<NNODES / 16, 256, 0, stream>>>(neigh, (const v16h*)Wp, b, h1);

    // 4. hop 2 aggregate (same CSR, shared weights)
    rmp_aggregate<<<NNODES, UNITS, 0, stream>>>(h1, deg, idx, neigh);

    // 5. x2 = swish(neigh @ W + b) -> d_out
    rmp_gemm_swish_wmma<<<NNODES / 16, 256, 0, stream>>>(neigh, (const v16h*)Wp, b, x2);
}
